// Model_3049426780225
// MI455X (gfx1250) — compile-verified, matrix operand order checked by reading, not run on hardware
//
#include <hip/hip_runtime.h>
#include <math.h>

// Problem dims (match reference)
#define HDIM   512
#define BDIM   512
#define TSTEPS 784
#define NCLASS 10
#define RNN_BLOCKS 64          // persistent blocks (must all be co-resident)
#define RNN_THREADS 256        // 8 waves (wave32)
#define LDS_STRIDE 520         // 512 bf16 + 8 pad -> 260 dwords/row, spreads banks

typedef __attribute__((ext_vector_type(16))) __bf16          v16bf;
typedef __attribute__((ext_vector_type(8)))  float           v8f;
typedef __attribute__((ext_vector_type(8)))  unsigned short  u16x8;
typedef __attribute__((ext_vector_type(16))) unsigned short  u16x16;

__device__ __forceinline__ unsigned short f32_to_bf16(float f) {
  unsigned int u = __builtin_bit_cast(unsigned int, f);
  u += 0x7FFFu + ((u >> 16) & 1u);            // round-to-nearest-even
  return (unsigned short)(u >> 16);
}
__device__ __forceinline__ float bf16_to_f32(unsigned short h) {
  unsigned int u = ((unsigned int)h) << 16;
  return __builtin_bit_cast(float, u);
}

// Branchless tanh on the native v_exp_f32 path:
// tanh(x) = 1 - 2/(exp(2x)+1); saturates correctly to +/-1 for large |x|.
__device__ __forceinline__ float fast_tanh(float x) {
  float e = __expf(2.0f * x);
  return 1.0f - 2.0f * __builtin_amdgcn_rcpf(e + 1.0f);
}

// Assemble a 16x32 bf16 WMMA operand fragment from two 16B loads.
// CDNA5 16-bit A/B layout: lane L -> row = L%16, k-half = 8*(L/16);
// VGPR0-3 hold K=[kh..kh+7], VGPR4-7 hold K=[kh+16..kh+23].
__device__ __forceinline__ v16bf pack_frag(u16x8 lo, u16x8 hi) {
  u16x16 full;
#pragma unroll
  for (int i = 0; i < 8; ++i) { full[i] = lo[i]; full[i + 8] = hi[i]; }
  return __builtin_bit_cast(v16bf, full);
}
__device__ __forceinline__ v16bf load_frag_g(const unsigned short* base) {
  return pack_frag(*(const u16x8*)(base), *(const u16x8*)(base + 16));
}

// ---------------------------------------------------------------- prep ----
__global__ void prep_kernel(const float* __restrict__ W_hh,
                            const float* __restrict__ h0,
                            unsigned short* __restrict__ Wbf,
                            unsigned short* __restrict__ hbuf0,
                            unsigned int* __restrict__ sync,
                            float* __restrict__ out) {
  const int idx = blockIdx.x * blockDim.x + threadIdx.x;
  if (idx < HDIM * HDIM) Wbf[idx]   = f32_to_bf16(W_hh[idx]);
  if (idx < BDIM * HDIM) hbuf0[idx] = f32_to_bf16(h0[idx]);
  if (idx == 0) {
    sync[0] = 0u;   // arrive counter
    sync[1] = 0u;   // generation
    out[0] = 0.0f;  // loss accumulator
    out[1] = 0.0f;  // correct accumulator
  }
}

// ---------------------------------------------------- persistent RNN ----
// Block b: ji = b & 31 (one 16-wide hidden-column group, W rows pinned in LDS),
//          bihalf = b >> 5 (half of the 32 batch tiles).
// Wave w:  computes tiles (bi0, ji) and (bi0+1, ji) with bi0 = bihalf*16 + 2w,
//          sharing one LDS-resident B fragment per k-step across both WMMAs.
__global__ void __launch_bounds__(RNN_THREADS, 1)
rnn_kernel(const float* __restrict__ inputs,          // B x T fp32
           const int*   __restrict__ order,           // T
           const float* __restrict__ W_ih,            // H x 1
           const float* __restrict__ b_ih,            // H
           const float* __restrict__ b_hh,            // H
           const unsigned short* __restrict__ Wbf,    // H x H bf16 (row j, contiguous k)
           unsigned short* __restrict__ hbuf0,        // B x H bf16
           unsigned short* __restrict__ hbuf1,        // B x H bf16
           unsigned int* __restrict__ sync) {
  __shared__ unsigned short Wlds[16 * LDS_STRIDE];    // 16.25 KB, bank-padded

  const int tid    = threadIdx.x;
  const int lane   = tid & 31;
  const int wvid   = tid >> 5;                        // 0..7
  const int lmod   = lane & 15;
  const int half8  = (lane >> 4) << 3;                // 0 or 8

  const int ji     = blockIdx.x & 31;                 // hidden tile (fixed)
  const int bihalf = blockIdx.x >> 5;                 // 0 or 1
  const int bi0    = bihalf * 16 + 2 * wvid;
  const int bi1    = bi0 + 1;

  // ---- one-time: stage this block's 16 W rows into LDS (B operand) ----
  // 16 rows x 512 bf16; thread copies 16B chunks.
  for (int e = tid * 8; e < 16 * HDIM; e += RNN_THREADS * 8) {
    const int row = e >> 9;            // /512
    const int col = e & (HDIM - 1);
    *(u16x8*)(&Wlds[row * LDS_STRIDE + col]) =
        *(const u16x8*)(&Wbf[(ji * 16 + row) * HDIM + col]);
  }
  __syncthreads();

  const unsigned short* bofs = &Wlds[lmod * LDS_STRIDE + half8];

  unsigned int* count = sync;
  unsigned int* gen   = sync + 1;

  const int   j    = ji * 16 + lmod;
  const float win  = W_ih[j];                         // W_ih[:,0]
  const float bias = b_ih[j] + b_hh[j];

  for (int t = 0; t < TSTEPS; ++t) {
    const unsigned short* src = (t & 1) ? hbuf1 : hbuf0;
    unsigned short*       dst = (t & 1) ? hbuf0 : hbuf1;
    const int col = order[t];          // xs[t][b] = inputs[b][order[t]]

    const unsigned short* aptr0 = src + (bi0 * 16 + lmod) * HDIM + half8;
    const unsigned short* aptr1 = src + (bi1 * 16 + lmod) * HDIM + half8;

    v8f acc0 = {};
    v8f acc1 = {};
#pragma unroll 4
    for (int k0 = 0; k0 < HDIM; k0 += 32) {
      v16bf a0 = load_frag_g(aptr0 + k0);
      v16bf a1 = load_frag_g(aptr1 + k0);
      v16bf b  = pack_frag(*(const u16x8*)(bofs + k0),
                           *(const u16x8*)(bofs + k0 + 16));   // ds_load_b128 x2
      acc0 = __builtin_amdgcn_wmma_f32_16x16x32_bf16(
          false, a0, false, b, (short)0, acc0, false, false);
      acc1 = __builtin_amdgcn_wmma_f32_16x16x32_bf16(
          false, a1, false, b, (short)0, acc1, false, false);
    }

    // Epilogue: h_next = tanh(acc + x*w_in + b_ih + b_hh), branchless tanh.
#pragma unroll
    for (int v = 0; v < 8; ++v) {
      const int   br0 = bi0 * 16 + half8 + v;        // C/D: M = v + 8*(lane>=16)
      const int   br1 = bi1 * 16 + half8 + v;
      const float x0  = inputs[br0 * TSTEPS + col];
      const float x1  = inputs[br1 * TSTEPS + col];
      dst[br0 * HDIM + j] = f32_to_bf16(fast_tanh(acc0[v] + x0 * win + bias));
      dst[br1 * HDIM + j] = f32_to_bf16(fast_tanh(acc1[v] + x1 * win + bias));
    }

    // -------- device-wide barrier (arrive counter + generation) --------
    __syncthreads();
    if (tid == 0) {
      __threadfence();  // make this block's dst stores device-visible
      unsigned int prev = __hip_atomic_fetch_add(count, 1u, __ATOMIC_ACQ_REL,
                                                 __HIP_MEMORY_SCOPE_AGENT);
      if (prev == (unsigned int)(RNN_BLOCKS - 1)) {
        __hip_atomic_store(count, 0u, __ATOMIC_RELAXED, __HIP_MEMORY_SCOPE_AGENT);
        __hip_atomic_fetch_add(gen, 1u, __ATOMIC_ACQ_REL, __HIP_MEMORY_SCOPE_AGENT);
      } else {
        const unsigned int target = (unsigned int)(t + 1);
        while (__hip_atomic_load(gen, __ATOMIC_ACQUIRE,
                                 __HIP_MEMORY_SCOPE_AGENT) < target) {
          __builtin_amdgcn_s_sleep(2);
        }
      }
    }
    __syncthreads();
  }
}

// ------------------------------------------- classifier + loss/correct ----
__global__ void cls_kernel(const unsigned short* __restrict__ hfin,  // B x H bf16
                           const float* __restrict__ W_lin,          // NCLASS x H
                           const float* __restrict__ b_lin,          // NCLASS
                           const int*   __restrict__ y,              // B
                           float* __restrict__ out) {
  const int b = blockIdx.x * blockDim.x + threadIdx.x;
  if (b >= BDIM) return;

  float logit[NCLASS];
#pragma unroll
  for (int c = 0; c < NCLASS; ++c) logit[c] = b_lin[c];

  for (int k = 0; k < HDIM; ++k) {
    const float hv = bf16_to_f32(hfin[b * HDIM + k]);
#pragma unroll
    for (int c = 0; c < NCLASS; ++c)
      logit[c] = fmaf(hv, W_lin[c * HDIM + k], logit[c]);
  }

  float m = logit[0];
  int pred = 0;
#pragma unroll
  for (int c = 1; c < NCLASS; ++c)
    if (logit[c] > m) { m = logit[c]; pred = c; }

  float s = 0.0f;
#pragma unroll
  for (int c = 0; c < NCLASS; ++c) s += expf(logit[c] - m);
  const float lse = m + logf(s);

  const int yb = y[b];
  const float logp = logit[yb] - lse;
  atomicAdd(&out[0], -logp / (float)BDIM);
  if (pred == yb) atomicAdd(&out[1], 1.0f);
}

// ------------------------------------------------------------- launch ----
extern "C" void kernel_launch(void* const* d_in, const int* in_sizes, int n_in,
                              void* d_out, int out_size, void* d_ws, size_t ws_size,
                              hipStream_t stream) {
  const float* inputs = (const float*)d_in[0];
  const int*   y      = (const int*)  d_in[1];
  const int*   order  = (const int*)  d_in[2];
  const float* h0     = (const float*)d_in[3];
  const float* W_ih   = (const float*)d_in[4];
  const float* W_hh   = (const float*)d_in[5];
  const float* b_ih   = (const float*)d_in[6];
  const float* b_hh   = (const float*)d_in[7];
  const float* W_lin  = (const float*)d_in[8];
  const float* b_lin  = (const float*)d_in[9];

  char* ws = (char*)d_ws;
  unsigned short* Wbf   = (unsigned short*)ws;                                   // 512 KB
  unsigned short* hbuf0 = (unsigned short*)(ws + (size_t)HDIM * HDIM * 2);       // 512 KB
  unsigned short* hbuf1 = (unsigned short*)(ws + (size_t)HDIM * HDIM * 2
                                               + (size_t)BDIM * HDIM * 2);      // 512 KB
  unsigned int*   sync  = (unsigned int*)(ws + (size_t)HDIM * HDIM * 2
                                             + 2 * (size_t)BDIM * HDIM * 2);
  float* out = (float*)d_out;

  // 1) convert weights/h0 to bf16, reset barrier + output accumulators
  prep_kernel<<<(HDIM * HDIM + 255) / 256, 256, 0, stream>>>(
      W_hh, h0, Wbf, hbuf0, sync, out);

  // 2) persistent RNN over 784 steps with device-wide barrier per step
  rnn_kernel<<<RNN_BLOCKS, RNN_THREADS, 0, stream>>>(
      inputs, order, W_ih, b_ih, b_hh, Wbf, hbuf0, hbuf1, sync);

  // 3) classifier + log-softmax loss + accuracy (TSTEPS even -> final h in hbuf0)
  cls_kernel<<<(BDIM + 255) / 256, 256, 0, stream>>>(
      hbuf0, W_lin, b_lin, y, out);
}
